// BinarySurrogateBlock_73753178407505
// MI455X (gfx1250) — matrix operand order repaired and checked by reading, not compiled, not run on hardware
//
#include <hip/hip_runtime.h>

// ---------------------------------------------------------------------------
// Problem: y[b,t,o] = sum_i x[b,t,i] * W[o,i],  W = sum_k 2^bits[k]*(pos-neg)/scale
// M = 8*2048 = 16384, N = 4096, K = 4096  -> 549 GFLOP GEMM, compute bound.
// Strategy: decode W_int exactly into bf16 (|W_int| <= 255 is exact in bf16),
// convert x to bf16, run a v_wmma_f32_16x16x32_bf16 GEMM with f32 accum,
// apply 1/scale in the f32 epilogue.
// ---------------------------------------------------------------------------

typedef __attribute__((ext_vector_type(8)))  __bf16 bf16x8;
typedef __attribute__((ext_vector_type(16))) __bf16 bf16x16;
typedef __attribute__((ext_vector_type(8)))  float  floatx8;

#define D_IN   4096
#define D_OUT  4096
#define M_TOT  16384   // B*T
#define NPLANE 8

// ---------------------------------------------------------------------------
// Kernel 1: decode signed bit-planes -> W_int as bf16 (exact).
// Process 4 weights per thread via dword loads of the byte masks.
// P = sum_k (pos_bytes << bit_k): each byte stays <= 255, no cross-byte carry.
// ---------------------------------------------------------------------------
__global__ __launch_bounds__(256) void decode_w_kernel(
    const unsigned int* __restrict__ pos, const unsigned int* __restrict__ neg,
    const int* __restrict__ bits, unsigned short* __restrict__ Wb)
{
    constexpr int OI4 = (D_OUT * D_IN) / 4;
    const int t = blockIdx.x * 256 + threadIdx.x;   // t < OI4
    unsigned int P = 0, Ng = 0;
#pragma unroll
    for (int k = 0; k < NPLANE; ++k) {
        const unsigned int s = (unsigned int)bits[k];   // 0..7
        P  += pos[(size_t)k * OI4 + t] << s;
        Ng += neg[(size_t)k * OI4 + t] << s;
    }
    unsigned long long out = 0ull;
#pragma unroll
    for (int j = 0; j < 4; ++j) {
        const int w = (int)((P >> (8 * j)) & 0xFFu) - (int)((Ng >> (8 * j)) & 0xFFu);
        const float f = (float)w;   // |w| <= 255 -> exact in bf16, truncate is exact
        const unsigned short h = (unsigned short)(__builtin_bit_cast(unsigned int, f) >> 16);
        out |= ((unsigned long long)h) << (16 * j);
    }
    *(unsigned long long*)(Wb + 4 * (size_t)t) = out;
}

// ---------------------------------------------------------------------------
// Kernel 2: x fp32 -> bf16 (round to nearest even), 4 elements per thread.
// ---------------------------------------------------------------------------
__device__ __forceinline__ unsigned short f32_to_bf16_rne(float f) {
    unsigned int u = __builtin_bit_cast(unsigned int, f);
    u += 0x7FFFu + ((u >> 16) & 1u);
    return (unsigned short)(u >> 16);
}

__global__ __launch_bounds__(256) void cvt_x_kernel(
    const float4* __restrict__ X, unsigned long long* __restrict__ Xb)
{
    const int t = blockIdx.x * 256 + threadIdx.x;   // t < M*K/4
    const float4 v = X[t];
    const unsigned long long r =
          (unsigned long long)f32_to_bf16_rne(v.x)
        | ((unsigned long long)f32_to_bf16_rne(v.y) << 16)
        | ((unsigned long long)f32_to_bf16_rne(v.z) << 32)
        | ((unsigned long long)f32_to_bf16_rne(v.w) << 48);
    Xb[t] = r;
}

// ---------------------------------------------------------------------------
// Kernel 3: bf16 WMMA GEMM.  y[m,n] = sum_k Xb[m,k]*Wb[n,k]  (B = W^T, so a
// "column" of B is a contiguous row segment of W in memory).
//
// Fragment loads match the CDNA5 wave32 VGPR layouts:
//  A (16x32 bf16): lane L: row = m0 + (L&15), kh = L>>4.
//    VGPR0-3 hold K = kh*8 + 0..7   (16 contiguous bytes)
//    VGPR4-7 hold K = kh*8 + 16..23 (16 contiguous bytes, +32B away)
//  B (32x16 bf16): lane L: col = n0 + (L&15), kh = L>>4.
//    VGPR0-3 hold K = kh*8 + 0..7, VGPR4-7 hold K = kh*8 + 16..23 (same as A)
//  C/D (16x16 f32): VGPR r: lanes0-15 -> M=r, lanes16-31 -> M=r+8; N = L&15.
// ---------------------------------------------------------------------------
__device__ __forceinline__ bf16x16 load_a_frag(const __bf16* p) {
    bf16x8 lo = *(const bf16x8*)(p);
    bf16x8 hi = *(const bf16x8*)(p + 16);
    return __builtin_shufflevector(lo, hi, 0,1,2,3,4,5,6,7,8,9,10,11,12,13,14,15);
}
__device__ __forceinline__ bf16x16 load_b_frag(const __bf16* p) {
    bf16x8 lo = *(const bf16x8*)(p);
    bf16x8 hi = *(const bf16x8*)(p + 16);
    return __builtin_shufflevector(lo, hi, 0,1,2,3,4,5,6,7,8,9,10,11,12,13,14,15);
}

#define WMMA_BF16(a, b, c) \
    __builtin_amdgcn_wmma_f32_16x16x32_bf16(false, (a), false, (b), (short)0, (c), false, false)

__global__ __launch_bounds__(256) void wmma_gemm_kernel(
    const __bf16* __restrict__ Xb, const __bf16* __restrict__ Wb,
    const float* __restrict__ scale_p, float* __restrict__ Y)
{
    constexpr int K = D_IN;
    constexpr int N = D_OUT;

    const int lane = threadIdx.x & 31;
    const int wave = threadIdx.x >> 5;
    const int wm   = wave & 1;          // 2 wave rows   (64 rows each)
    const int wn   = wave >> 1;         // 4 wave cols   (32 cols each)
    const int l15  = lane & 15;
    const int kh   = lane >> 4;

    const int mWave = blockIdx.y * 128 + wm * 64;
    const int nWave = blockIdx.x * 128 + wn * 32;

    const __bf16* aPtr[4];
#pragma unroll
    for (int mt = 0; mt < 4; ++mt)
        aPtr[mt] = Xb + (size_t)(mWave + mt * 16 + l15) * K + kh * 8;
    const __bf16* bPtr[2];
#pragma unroll
    for (int nt = 0; nt < 2; ++nt)
        bPtr[nt] = Wb + (size_t)(nWave + nt * 16 + l15) * K + kh * 8;

    floatx8 acc[4][2];
#pragma unroll
    for (int mt = 0; mt < 4; ++mt)
#pragma unroll
        for (int nt = 0; nt < 2; ++nt)
            acc[mt][nt] = (floatx8){0.f,0.f,0.f,0.f,0.f,0.f,0.f,0.f};

    // Double-buffered K loop, 2 K-steps (64) per iteration; all buffer indices
    // are compile-time constants so fragments stay in VGPRs.
    bf16x16 a0[4], b0[2], a1[4], b1[2];
#pragma unroll
    for (int mt = 0; mt < 4; ++mt) a0[mt] = load_a_frag(aPtr[mt]);
#pragma unroll
    for (int nt = 0; nt < 2; ++nt) b0[nt] = load_b_frag(bPtr[nt]);

    for (int k0 = 0; k0 < K; k0 += 64) {
        // prefetch k0+32 while computing on k0
#pragma unroll
        for (int mt = 0; mt < 4; ++mt) a1[mt] = load_a_frag(aPtr[mt] + k0 + 32);
#pragma unroll
        for (int nt = 0; nt < 2; ++nt) b1[nt] = load_b_frag(bPtr[nt] + k0 + 32);
#pragma unroll
        for (int mt = 0; mt < 4; ++mt)
#pragma unroll
            for (int nt = 0; nt < 2; ++nt)
                acc[mt][nt] = WMMA_BF16(a0[mt], b0[nt], acc[mt][nt]);

        if (k0 + 64 < K) {
#pragma unroll
            for (int mt = 0; mt < 4; ++mt) a0[mt] = load_a_frag(aPtr[mt] + k0 + 64);
#pragma unroll
            for (int nt = 0; nt < 2; ++nt) b0[nt] = load_b_frag(bPtr[nt] + k0 + 64);
        }
#pragma unroll
        for (int mt = 0; mt < 4; ++mt)
#pragma unroll
            for (int nt = 0; nt < 2; ++nt)
                acc[mt][nt] = WMMA_BF16(a1[mt], b1[nt], acc[mt][nt]);
    }

    const float inv = 1.0f / scale_p[0];
#pragma unroll
    for (int mt = 0; mt < 4; ++mt)
#pragma unroll
        for (int nt = 0; nt < 2; ++nt) {
            const int col  = nWave + nt * 16 + l15;
            const int row0 = mWave + mt * 16 + kh * 8;
            float* yp = Y + (size_t)row0 * N + col;
#pragma unroll
            for (int r = 0; r < 8; ++r)
                yp[(size_t)r * N] = acc[mt][nt][r] * inv;
        }
}

// ---------------------------------------------------------------------------
// Host-side launcher. Inputs: x(f32), pos_masks(u8), neg_masks(u8),
// bits(i32), scale(f32 scalar). Workspace: [0,32MB) W bf16, [32MB,160MB) x bf16.
// ---------------------------------------------------------------------------
extern "C" void kernel_launch(void* const* d_in, const int* in_sizes, int n_in,
                              void* d_out, int out_size, void* d_ws, size_t ws_size,
                              hipStream_t stream) {
    const float*        x     = (const float*)d_in[0];
    const unsigned int* pos   = (const unsigned int*)d_in[1];  // bool bytes, dword view
    const unsigned int* neg   = (const unsigned int*)d_in[2];
    const int*          bits  = (const int*)d_in[3];
    const float*        scale = (const float*)d_in[4];
    float*              y     = (float*)d_out;

    __bf16* Wb = (__bf16*)d_ws;
    __bf16* Xb = (__bf16*)((char*)d_ws + (size_t)D_OUT * D_IN * 2);

    // 1) decode W bit-planes -> bf16 (exact integers)
    decode_w_kernel<<<(D_OUT * D_IN / 4) / 256, 256, 0, stream>>>(
        pos, neg, bits, (unsigned short*)Wb);

    // 2) x fp32 -> bf16 (RNE)
    cvt_x_kernel<<<((size_t)M_TOT * D_IN / 4) / 256, 256, 0, stream>>>(
        (const float4*)x, (unsigned long long*)Xb);

    // 3) bf16 WMMA GEMM + 1/scale epilogue
    wmma_gemm_kernel<<<dim3(D_OUT / 128, M_TOT / 128), 256, 0, stream>>>(
        Xb, Wb, scale, y);
}
